// LIIF_90142773609246
// MI455X (gfx1250) — matrix operand order repaired and checked by reading, not compiled, not run on hardware
//
#include <hip/hip_runtime.h>
#include <hip/hip_bf16.h>
#include <hip/hip_fp16.h>

#define BB 4
#define CC 64
#define HH 128
#define WW 128
#define QQ 30000
#define HID 256
#define INDIM 580
#define KPAD 608
#define UDIM 576   // C*9 unfolded features; 18 full 32-K chunks

typedef _Float16 h8  __attribute__((ext_vector_type(8)));
typedef _Float16 h16 __attribute__((ext_vector_type(16)));
typedef float    f8  __attribute__((ext_vector_type(8)));

union H16U { h16 v; h8 g[2]; };

__device__ __forceinline__ f8 wmma_f16(h16 a, h16 b, f8 c) {
  // D = A(16x32 f16) * B(32x16 f16) + C(16x16 f32)
  return __builtin_amdgcn_wmma_f32_16x16x32_f16(false, a, false, b, (short)0, c,
                                                false, false);
}

// ---------------- conv3x3 (1 -> 64 ch), zero pad, output f16 channel-last ----
__global__ void k_conv(const float* __restrict__ inp, const float* __restrict__ cw,
                       const float* __restrict__ cb, _Float16* __restrict__ feat) {
  __shared__ float sw[CC * 9];
  __shared__ float sb[CC];
  for (int t = threadIdx.x; t < CC * 9; t += blockDim.x) sw[t] = cw[t];
  for (int t = threadIdx.x; t < CC; t += blockDim.x) sb[t] = cb[t];
  __syncthreads();
  int p = blockIdx.x * blockDim.x + threadIdx.x;
  if (p >= BB * HH * WW) return;
  int w = p % WW, h = (p / WW) % HH, b = p / (HH * WW);
  float pix[9];
#pragma unroll
  for (int kh = 0; kh < 3; ++kh)
#pragma unroll
    for (int kw = 0; kw < 3; ++kw) {
      int hh = h + kh - 1, ww2 = w + kw - 1;
      pix[kh * 3 + kw] = (hh >= 0 && hh < HH && ww2 >= 0 && ww2 < WW)
                             ? inp[((size_t)b * HH + hh) * WW + ww2]
                             : 0.f;
    }
  _Float16* o = feat + (size_t)p * CC;
  for (int c = 0; c < CC; ++c) {
    float s = sb[c];
#pragma unroll
    for (int t = 0; t < 9; ++t) s += sw[c * 9 + t] * pix[t];
    o[c] = (_Float16)s;
  }
}

// ------- unfold: featu[b][pix][c*9+tap] = feat[b][pix(dh,dw)][c] (zero pad) --
// Makes each query's 576-dim feature a contiguous, 16B-aligned row.
__global__ void k_unfold(const _Float16* __restrict__ feat,
                         _Float16* __restrict__ featu) {
  int t = blockIdx.x * blockDim.x + threadIdx.x;  // pixel*64 + c
  if (t >= BB * HH * WW * CC) return;
  int c = t & 63;
  int pix = t >> 6;
  int w = pix % WW, h = (pix / WW) % HH, b = pix / (HH * WW);
  _Float16* dst = featu + (size_t)pix * UDIM + c * 9;
#pragma unroll
  for (int kh = 0; kh < 3; ++kh)
#pragma unroll
    for (int kw = 0; kw < 3; ++kw) {
      int ii = h + kh - 1, jj = w + kw - 1;
      _Float16 v = (_Float16)0.f;
      if (ii >= 0 && ii < HH && jj >= 0 && jj < WW)
        v = feat[((size_t)(b * HH + ii) * WW + jj) * CC + c];
      dst[kh * 3 + kw] = v;
    }
}

// ------------- weight prep: transpose to WT[n][k], pad K and N, f32->f16 -----
__global__ void k_wprep(const float* __restrict__ w, _Float16* __restrict__ wt,
                        int K, int N, int Kpad, int Npad) {
  int idx = blockIdx.x * blockDim.x + threadIdx.x;
  if (idx >= Npad * Kpad) return;
  int n = idx / Kpad, k = idx % Kpad;
  float v = (k < K && n < N) ? w[(size_t)k * N + n] : 0.f;
  wt[idx] = (_Float16)v;
}

// ---- hidden MLP layer (256->256), TWO 16-row M-tiles sharing each B frag ---
// A (ISA 16-bit A 16x32): lane -> row M=lane&15; two 8-half runs per 32-K
// chunk at (lane>>4)*8 and +16.   B (f16 32x16): lane -> col N=lane&15,
// same K order as A: two 8-half runs at (lane>>4)*8 and +16 of each chunk.
__device__ __forceinline__ void mlp_layer2(
    const _Float16* __restrict__ Ain,
    const _Float16* __restrict__ Wt,
    const float* __restrict__ bias,
    _Float16* __restrict__ Hout, int lane) {
  const int ln = lane & 15;
  const int hb = lane >> 4;
  for (int n0 = 0; n0 < 16; ++n0) {
    f8 acc0 = {}, acc1 = {};
    const _Float16* __restrict__ wrow = Wt + (size_t)(n0 * 16 + ln) * HID + (hb << 3);
    const _Float16* __restrict__ arow0 = Ain + ln * HID + (hb << 3);
    const _Float16* __restrict__ arow1 = arow0 + 16 * HID;
    for (int kc = 0; kc < HID / 32; ++kc) {
      H16U bf, a0, a1;
      bf.g[0] = *(const h8*)(wrow + kc * 32);
      bf.g[1] = *(const h8*)(wrow + kc * 32 + 16);
      a0.g[0] = *(const h8*)(arow0 + kc * 32);
      a0.g[1] = *(const h8*)(arow0 + kc * 32 + 16);
      a1.g[0] = *(const h8*)(arow1 + kc * 32);
      a1.g[1] = *(const h8*)(arow1 + kc * 32 + 16);
      acc0 = wmma_f16(a0.v, bf.v, acc0);   // B reused by both tiles
      acc1 = wmma_f16(a1.v, bf.v, acc1);
    }
    const float bv = bias[n0 * 16 + ln];
    // D: M = r + 8*(lane>>4), N = lane&15
#pragma unroll
    for (int rI = 0; rI < 8; ++rI) {
      float v0 = fmaxf(acc0[rI] + bv, 0.f);
      float v1 = fmaxf(acc1[rI] + bv, 0.f);
      Hout[(rI + (hb << 3)) * HID + n0 * 16 + ln] = (_Float16)v0;
      Hout[(16 + rI + (hb << 3)) * HID + n0 * 16 + ln] = (_Float16)v1;
    }
  }
}

// ------------------------- main fused MLP kernel -----------------------------
// One wave per block; each wave owns 32 rows (2 WMMA M-tiles).
// Layer-1 A fragments stream DIRECTLY from global featu (per-lane row base);
// the 4 rel/cell tail values are synthesized in registers (K-chunk 18).
__global__ void __launch_bounds__(32, 2) k_mlp(
    const _Float16* __restrict__ featu,
    const float* __restrict__ coord, const float* __restrict__ cell,
    const _Float16* __restrict__ w1t, const _Float16* __restrict__ w2t,
    const _Float16* __restrict__ w3t, const _Float16* __restrict__ w4t,
    const _Float16* __restrict__ w5t,
    const float* __restrict__ b1, const float* __restrict__ b2,
    const float* __restrict__ b3, const float* __restrict__ b4,
    const float* __restrict__ b5,
    float* __restrict__ pred) {
  __shared__ __align__(16) _Float16 Ha[32 * HID];   // 16 KB
  __shared__ __align__(16) _Float16 Hc[32 * HID];   // 16 KB
  const int lane = threadIdx.x & 31;
  const int ln = lane & 15;
  const int hb = lane >> 4;
  const int rowBase = blockIdx.x * 32;

  // ---- per-lane decode: this lane owns row (rowBase + t*16 + ln) of tile t --
  const _Float16* abase[2];
  H16U tail[2];
#pragma unroll
  for (int t = 0; t < 2; ++t) {
    const int row = rowBase + t * 16 + ln;
    const int br = row & 3;
    const int bq = row >> 2;  // b*Q + q
    const int b = bq / QQ;
    const float c0 = coord[bq * 2 + 0];
    const float c1 = coord[bq * 2 + 1];
    const float vx = (br & 2) ? 1.f : -1.f;
    const float vy = (br & 1) ? 1.f : -1.f;
    const float cc0 = fminf(fmaxf(c0 + vx * (1.f / HH) + 1e-6f, -1.f + 1e-6f), 1.f - 1e-6f);
    const float cc1 = fminf(fmaxf(c1 + vy * (1.f / WW) + 1e-6f, -1.f + 1e-6f), 1.f - 1e-6f);
    const float fi = fminf(fmaxf(rintf((cc0 + 1.f) * (HH * 0.5f) - 0.5f), 0.f), (float)(HH - 1));
    const float fj = fminf(fmaxf(rintf((cc1 + 1.f) * (WW * 0.5f) - 0.5f), 0.f), (float)(WW - 1));
    const int i = (int)fi, j = (int)fj;
    const float qy = -1.f + (2.f * fi + 1.f) / HH;
    const float qx = -1.f + (2.f * fj + 1.f) / WW;
    const float rel0 = (c0 - qy) * HH;
    const float rel1 = (c1 - qx) * WW;
    const float rc0 = cell[bq * 2 + 0] * HH;
    const float rc1 = cell[bq * 2 + 1] * WW;
    abase[t] = featu + ((size_t)b * (HH * WW) + i * WW + j) * UDIM + (hb << 3);
    // tail K-chunk (576..607): hb=0 lanes hold K=576..583,592..599
#pragma unroll
    for (int z = 0; z < 16; ++z) tail[t].v[z] = (_Float16)0.f;
    tail[t].v[0] = (hb == 0) ? (_Float16)rel0 : (_Float16)0.f;
    tail[t].v[1] = (hb == 0) ? (_Float16)rel1 : (_Float16)0.f;
    tail[t].v[2] = (hb == 0) ? (_Float16)rc0 : (_Float16)0.f;
    tail[t].v[3] = (hb == 0) ? (_Float16)rc1 : (_Float16)0.f;
  }

  // ---- layer 1: 580 -> 256, A from global featu + register tail ----
  for (int n0 = 0; n0 < 16; ++n0) {
    f8 acc0 = {}, acc1 = {};
    const _Float16* __restrict__ wrow = w1t + (size_t)(n0 * 16 + ln) * KPAD + (hb << 3);
    for (int kc = 0; kc < UDIM / 32; ++kc) {  // 18 full chunks cover K=0..575
      H16U bf, a0, a1;
      bf.g[0] = *(const h8*)(wrow + kc * 32);
      bf.g[1] = *(const h8*)(wrow + kc * 32 + 16);
      a0.g[0] = *(const h8*)(abase[0] + kc * 32);
      a0.g[1] = *(const h8*)(abase[0] + kc * 32 + 16);
      a1.g[0] = *(const h8*)(abase[1] + kc * 32);
      a1.g[1] = *(const h8*)(abase[1] + kc * 32 + 16);
      acc0 = wmma_f16(a0.v, bf.v, acc0);
      acc1 = wmma_f16(a1.v, bf.v, acc1);
    }
    {  // tail chunk kc=18 (K=576..607), A synthesized in registers
      H16U bf;
      bf.g[0] = *(const h8*)(wrow + 18 * 32);
      bf.g[1] = *(const h8*)(wrow + 18 * 32 + 16);
      acc0 = wmma_f16(tail[0].v, bf.v, acc0);
      acc1 = wmma_f16(tail[1].v, bf.v, acc1);
    }
    const float bv = b1[n0 * 16 + ln];
#pragma unroll
    for (int rI = 0; rI < 8; ++rI) {
      float v0 = fmaxf(acc0[rI] + bv, 0.f);
      float v1 = fmaxf(acc1[rI] + bv, 0.f);
      Ha[(rI + (hb << 3)) * HID + n0 * 16 + ln] = (_Float16)v0;
      Ha[(16 + rI + (hb << 3)) * HID + n0 * 16 + ln] = (_Float16)v1;
    }
  }
  // LDS ops are in-order within a wave: no barriers needed (single wave).

  mlp_layer2(Ha, w2t, b2, Hc, lane);  // 256->256
  mlp_layer2(Hc, w3t, b3, Ha, lane);
  mlp_layer2(Ha, w4t, b4, Hc, lane);

  // ---- layer 5: 256 -> 1 (w5 padded to 16 cols; only col 0 real) ----
  {
    f8 acc0 = {}, acc1 = {};
    const _Float16* wrow = w5t + (size_t)ln * HID + (hb << 3);
    const _Float16* arow0 = Hc + ln * HID + (hb << 3);
    const _Float16* arow1 = arow0 + 16 * HID;
    for (int kc = 0; kc < HID / 32; ++kc) {
      H16U bf, a0, a1;
      bf.g[0] = *(const h8*)(wrow + kc * 32);
      bf.g[1] = *(const h8*)(wrow + kc * 32 + 16);
      a0.g[0] = *(const h8*)(arow0 + kc * 32);
      a0.g[1] = *(const h8*)(arow0 + kc * 32 + 16);
      a1.g[0] = *(const h8*)(arow1 + kc * 32);
      a1.g[1] = *(const h8*)(arow1 + kc * 32 + 16);
      acc0 = wmma_f16(a0.v, bf.v, acc0);
      acc1 = wmma_f16(a1.v, bf.v, acc1);
    }
    if (ln == 0) {  // lanes 0 and 16 hold column N=0: M = r + 8*(lane>>4)
      const float b5v = b5[0];
#pragma unroll
      for (int rI = 0; rI < 8; ++rI) {
        pred[rowBase + rI + (hb << 3)] = acc0[rI] + b5v;
        pred[rowBase + 16 + rI + (hb << 3)] = acc1[rI] + b5v;
      }
    }
  }
}

// ------------------- local-ensemble combine (areas recomputed) ---------------
__global__ void k_combine(const float* __restrict__ pred,
                          const float* __restrict__ coord,
                          const float* __restrict__ cell,
                          float* __restrict__ out) {
  int p = blockIdx.x * blockDim.x + threadIdx.x;
  if (p >= BB * QQ) return;
  const float c0 = coord[p * 2 + 0], c1 = coord[p * 2 + 1];
  float areas[4], pr[4];
#pragma unroll
  for (int br = 0; br < 4; ++br) {
    const float vx = (br & 2) ? 1.f : -1.f;
    const float vy = (br & 1) ? 1.f : -1.f;
    const float cc0 = fminf(fmaxf(c0 + vx * (1.f / HH) + 1e-6f, -1.f + 1e-6f), 1.f - 1e-6f);
    const float cc1 = fminf(fmaxf(c1 + vy * (1.f / WW) + 1e-6f, -1.f + 1e-6f), 1.f - 1e-6f);
    const float fi = fminf(fmaxf(rintf((cc0 + 1.f) * (HH * 0.5f) - 0.5f), 0.f), (float)(HH - 1));
    const float fj = fminf(fmaxf(rintf((cc1 + 1.f) * (WW * 0.5f) - 0.5f), 0.f), (float)(WW - 1));
    const float qy = -1.f + (2.f * fi + 1.f) / HH;
    const float qx = -1.f + (2.f * fj + 1.f) / WW;
    const float rel0 = (c0 - qy) * HH;
    const float rel1 = (c1 - qx) * WW;
    areas[br] = fabsf(rel0 * rel1) + 1e-9f;
    pr[br] = pred[p * 4 + br];
  }
  const float tot = areas[0] + areas[1] + areas[2] + areas[3];
  float r = 0.f;
#pragma unroll
  for (int t = 0; t < 4; ++t) r += pr[t] * (areas[3 - t] / tot);  // diagonal swap
  out[p] = r;
}

extern "C" void kernel_launch(void* const* d_in, const int* in_sizes, int n_in,
                              void* d_out, int out_size, void* d_ws, size_t ws_size,
                              hipStream_t stream) {
  const float* inp    = (const float*)d_in[0];
  const float* coord  = (const float*)d_in[1];
  const float* cell   = (const float*)d_in[2];
  const float* conv_w = (const float*)d_in[3];
  const float* conv_b = (const float*)d_in[4];
  const float* w1 = (const float*)d_in[5];  const float* b1 = (const float*)d_in[6];
  const float* w2 = (const float*)d_in[7];  const float* b2 = (const float*)d_in[8];
  const float* w3 = (const float*)d_in[9];  const float* b3 = (const float*)d_in[10];
  const float* w4 = (const float*)d_in[11]; const float* b4 = (const float*)d_in[12];
  const float* w5 = (const float*)d_in[13]; const float* b5 = (const float*)d_in[14];

  char* ws = (char*)d_ws;
  _Float16* feat  = (_Float16*)(ws + 0);          //  8,388,608 B
  _Float16* featu = (_Float16*)(ws + 8388608);    // 75,497,472 B (L2-resident)
  _Float16* w1t   = (_Float16*)(ws + 83886080);   //    311,296 B
  _Float16* w2t   = (_Float16*)(ws + 84197376);   //    131,072 B
  _Float16* w3t   = (_Float16*)(ws + 84328448);
  _Float16* w4t   = (_Float16*)(ws + 84459520);
  _Float16* w5t   = (_Float16*)(ws + 84590592);   //      8,192 B
  float*    pred  = (float*)(ws + 84598784);      //  1,920,000 B

  k_conv<<<(BB * HH * WW + 255) / 256, 256, 0, stream>>>(inp, conv_w, conv_b, feat);
  k_unfold<<<(BB * HH * WW * CC + 255) / 256, 256, 0, stream>>>(feat, featu);
  k_wprep<<<(HID * KPAD + 255) / 256, 256, 0, stream>>>(w1, w1t, INDIM, HID, KPAD, HID);
  k_wprep<<<(HID * HID + 255) / 256, 256, 0, stream>>>(w2, w2t, HID, HID, HID, HID);
  k_wprep<<<(HID * HID + 255) / 256, 256, 0, stream>>>(w3, w3t, HID, HID, HID, HID);
  k_wprep<<<(HID * HID + 255) / 256, 256, 0, stream>>>(w4, w4t, HID, HID, HID, HID);
  k_wprep<<<(16 * HID + 255) / 256, 256, 0, stream>>>(w5, w5t, HID, 1, HID, 16);

  // 480000 rows / 32 rows-per-wave = 15000 single-wave blocks
  k_mlp<<<15000, 32, 0, stream>>>(featu, coord, cell, w1t, w2t, w3t, w4t, w5t,
                                  b1, b2, b3, b4, b5, pred);
  k_combine<<<(BB * QQ + 255) / 256, 256, 0, stream>>>(pred, coord, cell, (float*)d_out);
}
